// CrossOutLayer_62646392980066
// MI455X (gfx1250) — matrix operand order repaired and checked by reading, not compiled, not run on hardware
//
#include <hip/hip_runtime.h>
#include <cmath>
#include <stdint.h>

typedef __attribute__((ext_vector_type(16))) _Float16 v16h;
typedef __attribute__((ext_vector_type(8)))  _Float16 v8h;
typedef __attribute__((ext_vector_type(8)))  float    v8f;

#define Dd 256
#define NN1 512
#define NN2 512
#define BB  4

// B-tile staged in LDS, transposed, f16: sB[col][k], col 0..63, k 0..255.
// Stride 264 halves = 528 B: 16B-aligned rows, banks spread by 4 dwords/col.
#define BSTRIDE 264

// ---------------------------------------------------------------------------
// Kernel 1: xp = x @ W1[:256] + b1   and   yp = y @ W1[256:]
// A: 2048x256 (row-major), W: 256x256 (row-major). f16 WMMA, f32 accumulate.
// W block (256x64) converted to f16 and transposed into LDS once per block;
// B fragments come from LDS as ds_load_b128 pairs. One wave per 16x16 tile.
// ---------------------------------------------------------------------------
__global__ void __launch_bounds__(128) gemm_wmma_kernel(
    const float* __restrict__ x, const float* __restrict__ y,
    const float* __restrict__ W1, const float* __restrict__ b1,
    float* __restrict__ xp, float* __restrict__ yp)
{
    const int z = blockIdx.z;                       // 0: x-path, 1: y-path
    const float* A    = (z == 0) ? x  : y;
    const float* W    = (z == 0) ? W1 : (W1 + Dd * Dd);
    float*       outp = (z == 0) ? xp : yp;

    __shared__ __attribute__((aligned(16))) _Float16 sB[64 * BSTRIDE];

    const int nb0 = blockIdx.x * 64;

    // Stage + transpose-convert W[k][nb0+c] -> sB[c*BSTRIDE + k] (f16).
    // Coalesced float4 global reads; scalar b16 LDS stores.
    for (int i = threadIdx.x; i < (Dd * 64) / 4; i += 128) {
        const int k  = i >> 4;                      // 0..255
        const int c4 = (i & 15) * 4;                // 0,4,...,60
        const float4 w4 = *(const float4*)(W + (size_t)k * Dd + nb0 + c4);
        sB[(c4 + 0) * BSTRIDE + k] = (_Float16)w4.x;
        sB[(c4 + 1) * BSTRIDE + k] = (_Float16)w4.y;
        sB[(c4 + 2) * BSTRIDE + k] = (_Float16)w4.z;
        sB[(c4 + 3) * BSTRIDE + k] = (_Float16)w4.w;
    }
    __syncthreads();

    const int wave  = threadIdx.x >> 5;
    const int lane  = threadIdx.x & 31;
    const int row16 = lane & 15;                    // row (A/D) or col (B/D)
    const int hi    = lane >> 4;                    // lane half-group

    const int mbase = blockIdx.y * 16;              // output row tile
    const int nbase = nb0 + wave * 16;              // output col tile

    union { v8f v; float e[8]; } c;
    c.v = (v8f)0.0f;

    const float*    Arow = A + (size_t)(mbase + row16) * Dd;
    const _Float16* bcol = sB + (size_t)(wave * 16 + row16) * BSTRIDE;

    for (int kb = 0; kb < Dd; kb += 32) {
        union { v16h v; v8h h[2]; _Float16 e[16]; } a, b;

        // A fragment: half e -> K = kb + (e>>3)*16 + hi*8 + (e&7)
        #pragma unroll
        for (int g = 0; g < 2; ++g) {
            const float* src = Arow + kb + g * 16 + hi * 8;
            #pragma unroll
            for (int w = 0; w < 8; ++w)
                a.e[g * 8 + w] = (_Float16)src[w];
        }
        // B fragment: half e -> K = kb + (e>>3)*16 + hi*8 + (e&7)
        b.h[0] = *(const v8h*)(bcol + kb + hi * 8);
        b.h[1] = *(const v8h*)(bcol + kb + 16 + hi * 8);

        c.v = __builtin_amdgcn_wmma_f32_16x16x32_f16(
                  false, a.v, false, b.v, (short)0, c.v, false, false);
    }

    // D layout: reg r, lane -> (M = mbase + r + 8*hi, N = nbase + (lane&15))
    const int   ncol = nbase + row16;
    const float bias = (z == 0) ? b1[ncol] : 0.0f;  // fold b1 into xp
    #pragma unroll
    for (int r = 0; r < 8; ++r) {
        const int m = mbase + r + 8 * hi;
        outp[(size_t)m * Dd + ncol] = c.e[r] + bias;
    }
}

// ---------------------------------------------------------------------------
// Kernel 2: out[b,n,m] = sum_d gelu(xp[b,n,d] + yp[b,m,d]) * W2[d] + b2
// 32(n) x 16(m) tile per block; xp/yp rows staged in LDS with CDNA5 async
// copies (global_load_async_to_lds_b128, ASYNCcnt). Stride 260 floats:
// 16B-aligned rows, banks spread. 256 threads, 2 outputs/thread.
// ---------------------------------------------------------------------------
#define TN 32
#define TM 16
#define LSTRIDE 260

__global__ void __launch_bounds__(256) cross_gelu_dot_kernel(
    const float* __restrict__ xp, const float* __restrict__ yp,
    const float* __restrict__ W2, const float* __restrict__ b2,
    float* __restrict__ out)
{
    __shared__ __attribute__((aligned(16))) float sX[TN * LSTRIDE];
    __shared__ __attribute__((aligned(16))) float sY[TM * LSTRIDE];
    __shared__ float sW[Dd];

    const int b   = blockIdx.z;
    const int n0  = blockIdx.y * TN;
    const int m0  = blockIdx.x * TM;
    const int tid = threadIdx.x;

    const float* xb = xp + ((size_t)b * NN1 + n0) * Dd;
    const float* yb = yp + ((size_t)b * NN2 + m0) * Dd;

    // Async-copy the x rows (32 x 256 f32, 16B chunks) into LDS.
    for (int i = tid; i < TN * (Dd / 4); i += 256) {
        const int r  = i >> 6;                      // row 0..31
        const int c4 = (i & 63) * 4;                // col 0,4,...,252
        const uint32_t loff =
            (uint32_t)(uintptr_t)&sX[r * LSTRIDE + c4];
        const unsigned long long ga =
            (unsigned long long)(uintptr_t)(xb + (size_t)r * Dd + c4);
        asm volatile("global_load_async_to_lds_b128 %0, %1, off"
                     :: "v"(loff), "v"(ga) : "memory");
    }
    // Async-copy the y rows (16 x 256 f32).
    for (int i = tid; i < TM * (Dd / 4); i += 256) {
        const int r  = i >> 6;
        const int c4 = (i & 63) * 4;
        const uint32_t loff =
            (uint32_t)(uintptr_t)&sY[r * LSTRIDE + c4];
        const unsigned long long ga =
            (unsigned long long)(uintptr_t)(yb + (size_t)r * Dd + c4);
        asm volatile("global_load_async_to_lds_b128 %0, %1, off"
                     :: "v"(loff), "v"(ga) : "memory");
    }
    if (tid < Dd) sW[tid] = W2[tid];                // tiny, regular path

    asm volatile("s_wait_asynccnt 0" ::: "memory"); // drain this wave's asyncs
    __syncthreads();                                // all waves' asyncs visible

    const int nl  = tid >> 3;        // 0..31
    const int mlb = (tid & 7) * 2;   // 0,2,...,14  -> two consecutive m
    const float* xr = sX + nl * LSTRIDE;
    const float* y0 = sY + mlb * LSTRIDE;
    const float* y1 = y0 + LSTRIDE;

    float acc0 = 0.0f, acc1 = 0.0f;
    #pragma unroll 4
    for (int d = 0; d < Dd; ++d) {
        const float xv = xr[d];
        const float w  = sW[d];
        const float h0 = xv + y0[d];
        const float h1 = xv + y1[d];
        // exact gelu: 0.5*h*(1+erf(h/sqrt(2)))
        const float g0 = 0.5f * h0 * (1.0f + erff(h0 * 0.70710678118654752f));
        const float g1 = 0.5f * h1 * (1.0f + erff(h1 * 0.70710678118654752f));
        acc0 = fmaf(g0, w, acc0);
        acc1 = fmaf(g1, w, acc1);
    }

    const float bb = b2[0];
    const size_t base = ((size_t)b * NN1 + (n0 + nl)) * NN2 + m0 + mlb;
    out[base]     = acc0 + bb;
    out[base + 1] = acc1 + bb;
}

// ---------------------------------------------------------------------------
extern "C" void kernel_launch(void* const* d_in, const int* in_sizes, int n_in,
                              void* d_out, int out_size, void* d_ws, size_t ws_size,
                              hipStream_t stream)
{
    const float* x  = (const float*)d_in[0];   // (4,512,256)
    const float* y  = (const float*)d_in[1];   // (4,512,256)
    const float* W1 = (const float*)d_in[2];   // (512,256)
    const float* b1 = (const float*)d_in[3];   // (256,)
    const float* W2 = (const float*)d_in[4];   // (256,1)
    const float* b2 = (const float*)d_in[5];   // (1,)
    float* out = (float*)d_out;                // (4,512,512)

    float* xp = (float*)d_ws;                  // 2048*256 f32 = 2 MB
    float* yp = xp + (size_t)BB * NN1 * Dd;    // 2048*256 f32 = 2 MB

    // Stage 1: two WMMA GEMMs (z=0: x@Wx+b1 -> xp, z=1: y@Wy -> yp)
    dim3 g1(Dd / 64, (BB * NN1) / 16, 2);      // (4, 128, 2)
    gemm_wmma_kernel<<<g1, 128, 0, stream>>>(x, y, W1, b1, xp, yp);

    // Stage 2: pairwise gelu + dot with W2
    dim3 g2(NN2 / TM, NN1 / TN, BB);           // (32, 16, 4)
    cross_gelu_dot_kernel<<<g2, 256, 0, stream>>>(xp, yp, W2, b2, out);
}
